// CausalSelfAttention_1563368095790
// MI455X (gfx1250) — matrix operand order repaired and checked by reading, not compiled, not run on hardware
//
#include <hip/hip_runtime.h>
#include <hip/hip_bf16.h>
#include <cstddef>
#include <cstdint>

// ---------- CDNA5 WMMA fragment types (wave32) ----------
typedef __attribute__((ext_vector_type(16))) __bf16 v16bf;
typedef __attribute__((ext_vector_type(8)))  __bf16 v8bf;
typedef __attribute__((ext_vector_type(8)))  float  v8f;
typedef __attribute__((ext_vector_type(4)))  unsigned int u32x4;
typedef __attribute__((ext_vector_type(4)))  int    i32x4;
typedef __attribute__((ext_vector_type(8)))  int    i32x8;

// Problem constants (from reference)
#define Sdim 2048
#define Bdim 2
#define H2   4096
#define HID  2048
#define NH   32
#define DH   128
#define O3   12288   // 3*H2
#define MROWS 4096   // S*B

#if defined(__has_builtin)
#if __has_builtin(__builtin_amdgcn_tensor_load_to_lds) && \
    __has_builtin(__builtin_amdgcn_s_wait_tensorcnt)
#define USE_TDM 1
#endif
#endif

// Pack two 8-elem halves into one 16-elem A/B fragment (register pairing, no VALU)
__device__ __forceinline__ v16bf pack_frag(v8bf lo, v8bf hi)
{
    return __builtin_shufflevector(lo, hi, 0, 1, 2, 3, 4, 5, 6, 7,
                                           8, 9, 10, 11, 12, 13, 14, 15);
}

#ifdef USE_TDM
// ---------------------------------------------------------------------------
// Tensor Data Mover: 2D tile (rows x cols of bf16) from global -> LDS.
// D# built per CDNA5 ISA (count=1, type=2,
// data_size=2B, pad/iterate/gather disabled, workgroup_mask=0).
// This toolchain exposes the 6-arg builtin:
//   (u32x4 g0, i32x8 g1, i32x4 g2, i32x4 g3, i32x8 g4, i32 cpol)
// ---------------------------------------------------------------------------
__device__ __forceinline__ void tdm_load_2d_bf16(
    const void* gptr, unsigned lds_byte_addr,
    int tile_cols, int tile_rows, long long row_stride_elems)
{
    const unsigned long long ga = (unsigned long long)(uintptr_t)gptr;
    u32x4 g0;
    g0[0] = 1u;                                     // count=1 (valid user D#)
    g0[1] = lds_byte_addr;                          // lds_addr [63:32]
    g0[2] = (unsigned)(ga & 0xffffffffu);           // global_addr[31:0]
    g0[3] = (unsigned)((ga >> 32) & 0x01ffffffu)    // global_addr[56:32]
          | (2u << 30);                             // type=2 ("image")
    i32x8 g1;
    g1[0] = (1 << 16);                              // data_size=1 -> 2 bytes
    g1[1] = (tile_cols & 0xffff) << 16;             // tensor_dim0[15:0]
    g1[2] = ((tile_cols >> 16) & 0xffff)            // tensor_dim0[31:16]
          | ((tile_rows & 0xffff) << 16);           // tensor_dim1[15:0]
    g1[3] = ((tile_rows >> 16) & 0xffff)            // tensor_dim1[31:16]
          | ((tile_cols & 0xffff) << 16);           // tile_dim0
    g1[4] = (tile_rows & 0xffff);                   // tile_dim1 (tile_dim2=0)
    g1[5] = (int)(row_stride_elems & 0xffffffffll); // tensor_dim0_stride[31:0]
    g1[6] = (int)((row_stride_elems >> 32) & 0xffffll); // stride[47:32]
    g1[7] = 0;
    const i32x4 z4 = {0, 0, 0, 0};                  // groups 2/3 unused (<=2D)
    const i32x8 z8 = {0, 0, 0, 0, 0, 0, 0, 0};
    __builtin_amdgcn_tensor_load_to_lds(g0, g1, z4, z4, z8, 0);
}
#endif

// ---------------------------------------------------------------------------
// fp32 -> bf16 conversion (grid-stride)
// ---------------------------------------------------------------------------
__global__ __launch_bounds__(256) void f32_to_bf16_kernel(
    const float* __restrict__ in, __bf16* __restrict__ out, size_t n)
{
    size_t i = (size_t)blockIdx.x * blockDim.x + threadIdx.x;
    size_t stride = (size_t)gridDim.x * blockDim.x;
    for (; i < n; i += stride) out[i] = (__bf16)in[i];
}

// ---------------------------------------------------------------------------
// bf16 WMMA GEMM:  C[M,N] = A[M,K] * W[N,K]^T   (nn.Linear weight layout)
// Block tile 128x128, 8 waves, each wave = 16(M) x 128(N) strip.
// K staged in LDS in tiles of 64 (2 WMMA k-steps per stage) via TDM.
// ---------------------------------------------------------------------------
template <bool OUT_BF16>
__global__ __launch_bounds__(256) void gemm_bf16_wmma(
    const __bf16* __restrict__ A,   // [M,K] row-major
    const __bf16* __restrict__ W,   // [N,K] row-major
    void* __restrict__ Cout,        // [M,N] bf16 or f32
    int M, int N, int K)
{
    constexpr int TM = 128, TN = 128, TK = 64;
    __shared__ alignas(128) __bf16 lsA[TM * TK];   // 16 KB
    __shared__ alignas(128) __bf16 lsB[TN * TK];   // 16 KB

    const int tid  = threadIdx.x;
    const int wave = tid >> 5;
    const int lane = tid & 31;
    const int half = (lane >> 4) & 1;
    const int ln   = lane & 15;
    const int mBase = blockIdx.y * TM;
    const int nBase = blockIdx.x * TN;

    v8f acc[8];
#pragma unroll
    for (int i = 0; i < 8; ++i) acc[i] = (v8f)(0.0f);

    for (int kb = 0; kb < K; kb += TK) {
        __syncthreads();    // all waves done with previous tiles
#ifdef USE_TDM
        if (wave == 0) {
            tdm_load_2d_bf16(A + (size_t)mBase * K + kb,
                             (unsigned)(uintptr_t)(void*)lsA, TK, TM, K);
            tdm_load_2d_bf16(W + (size_t)nBase * K + kb,
                             (unsigned)(uintptr_t)(void*)lsB, TK, TN, K);
            __builtin_amdgcn_s_wait_tensorcnt(0);
        }
#else
        // ---- stage A (128x64) and B (128x64) tiles: 512 chunks of 32B each
#pragma unroll
        for (int c = tid; c < 512; c += 256) {
            const int r  = c >> 2;
            const int cc = c & 3;
            const v16bf va = *(const v16bf*)(A + (size_t)(mBase + r) * K + kb + cc * 16);
            *(v16bf*)(lsA + r * TK + cc * 16) = va;
            const v16bf vb = *(const v16bf*)(W + (size_t)(nBase + r) * K + kb + cc * 16);
            *(v16bf*)(lsB + r * TK + cc * 16) = vb;
        }
#endif
        // hint next K-tile toward L2 (global_prefetch_b8)
        if (kb + TK < K) {
            __builtin_prefetch(A + (size_t)(mBase + (tid >> 1)) * K + kb + TK, 0, 0);
            __builtin_prefetch(W + (size_t)(nBase + (tid >> 1)) * K + kb + TK, 0, 0);
        }
        __syncthreads();

#pragma unroll
        for (int ks = 0; ks < 2; ++ks) {
            // A fragment: lane holds row m=ln; K elems {16*half..16*half+15}
            const __bf16* arow = lsA + (wave * 16 + ln) * TK + ks * 32;
            const v16bf afrag = pack_frag(*(const v8bf*)(arow + 16 * half),
                                          *(const v8bf*)(arow + 16 * half + 8));
            // batch all B fragments first so ds_loads issue back-to-back
            v16bf bfrag[8];
#pragma unroll
            for (int nt = 0; nt < 8; ++nt)
                bfrag[nt] = *(const v16bf*)(lsB + (nt * 16 + ln) * TK + ks * 32 + 16 * half);
#pragma unroll
            for (int nt = 0; nt < 8; ++nt)
                acc[nt] = __builtin_amdgcn_wmma_f32_16x16x32_bf16(
                    false, afrag, false, bfrag[nt], (short)0, acc[nt], false, false);
        }
    }

    // ---- epilogue: C layout is (m = j + 8*half, n = ln) per 16x16 tile
#pragma unroll
    for (int nt = 0; nt < 8; ++nt) {
#pragma unroll
        for (int j = 0; j < 8; ++j) {
            const size_t mg = (size_t)(mBase + wave * 16 + j + 8 * half);
            const size_t ng = (size_t)(nBase + nt * 16 + ln);
            if (OUT_BF16) ((__bf16*)Cout)[mg * (size_t)N + ng] = (__bf16)acc[nt][j];
            else          ((float*)Cout)[mg * (size_t)N + ng]  = acc[nt][j];
        }
    }
}

// ---------------------------------------------------------------------------
// Flash-style causal attention. grid = (S/128, NH, B), 256 threads (8 waves).
// Each wave owns a 16-row strip of the 128-row M block.
// LDS: K tile (32KB, via TDM; reused for P after score phase) + V^T (32KB).
// ---------------------------------------------------------------------------
__global__ __launch_bounds__(256) void attn_flash(
    const __bf16* __restrict__ qkv,  // [S*B, 12288]: q | k | v
    __bf16* __restrict__ ctx)        // [S*B, 4096]
{
    __shared__ alignas(128) __bf16 lsK[128 * 128];   // K tile, then P staging
    __shared__ alignas(128) __bf16 lsVt[128 * 128];  // V transposed [dh][kv]

    const int tid  = threadIdx.x;
    const int wave = tid >> 5;
    const int lane = tid & 31;
    const int half = (lane >> 4) & 1;
    const int ln   = lane & 15;

    const int mb = blockIdx.x;            // row block
    const int h  = blockIdx.y;            // head
    const int b  = blockIdx.z;            // batch
    const float scale = 0.08838834764831845f;  // 1/sqrt(128)

    const int rowm = mb * 128 + wave * 16;     // wave strip base (seq index)
    // Q global row for A-fragment (A rows are m = lane%16)
    const size_t qrow = ((size_t)(rowm + ln) * Bdim + b) * O3 + (size_t)h * DH;

    float mrow[8], lrow[8];
    v8f   acc[8];
#pragma unroll
    for (int j = 0; j < 8; ++j) { mrow[j] = -3.0e38f; lrow[j] = 0.0f; acc[j] = (v8f)(0.0f); }

    for (int jb = 0; jb <= mb; ++jb) {
        __syncthreads();   // previous iteration done with lsK(P)/lsVt
#ifdef USE_TDM
        // K tile: 2D tensor, 128 rows x 128 cols, row stride B*O3 elems
        if (wave == 0)
            tdm_load_2d_bf16(qkv + ((size_t)(jb * 128) * Bdim + b) * O3 + H2 + (size_t)h * DH,
                             (unsigned)(uintptr_t)(void*)lsK, 128, 128,
                             (long long)Bdim * O3);
        // V tile transposed by all threads (TDM cannot transpose) - overlaps TDM
#pragma unroll
        for (int c = tid; c < 1024; c += 256) {
            const int kv = c >> 3;
            const int ch = c & 7;
            const size_t soff = ((size_t)(jb * 128 + kv) * Bdim + b) * O3 + (size_t)h * DH;
            const v16bf vv = *(const v16bf*)(qkv + soff + 2 * H2 + ch * 16);
#pragma unroll
            for (int i = 0; i < 16; ++i) lsVt[(ch * 16 + i) * 128 + kv] = vv[i];
        }
        if (wave == 0) __builtin_amdgcn_s_wait_tensorcnt(0);
#else
        // ---- stage K tile row-major, V tile transposed
#pragma unroll
        for (int c = tid; c < 1024; c += 256) {
            const int kv = c >> 3;
            const int ch = c & 7;
            const size_t soff = ((size_t)(jb * 128 + kv) * Bdim + b) * O3 + (size_t)h * DH;
            const v16bf kk = *(const v16bf*)(qkv + soff + H2 + ch * 16);
            *(v16bf*)(lsK + kv * 128 + ch * 16) = kk;
            const v16bf vv = *(const v16bf*)(qkv + soff + 2 * H2 + ch * 16);
#pragma unroll
            for (int i = 0; i < 16; ++i) lsVt[(ch * 16 + i) * 128 + kv] = vv[i];
        }
#endif
        __syncthreads();

        // ---- S = Q * K^T  (8 column tiles x 4 k-steps of 32 over DH)
        v8f sacc[8];
#pragma unroll
        for (int nt = 0; nt < 8; ++nt) sacc[nt] = (v8f)(0.0f);
#pragma unroll
        for (int kc = 0; kc < 4; ++kc) {
            const v16bf aq = pack_frag(
                *(const v8bf*)(qkv + qrow + kc * 32 + 16 * half),
                *(const v8bf*)(qkv + qrow + kc * 32 + 16 * half + 8));
            v16bf bk[8];
#pragma unroll
            for (int nt = 0; nt < 8; ++nt)
                bk[nt] = *(const v16bf*)(lsK + (nt * 16 + ln) * 128 + kc * 32 + 16 * half);
#pragma unroll
            for (int nt = 0; nt < 8; ++nt)
                sacc[nt] = __builtin_amdgcn_wmma_f32_16x16x32_bf16(
                    false, aq, false, bk[nt], (short)0, sacc[nt], false, false);
        }

        // ---- scale + causal mask + online softmax update
        float mnew[8];
#pragma unroll
        for (int j = 0; j < 8; ++j) mnew[j] = mrow[j];
#pragma unroll
        for (int nt = 0; nt < 8; ++nt) {
            const int cgl = jb * 128 + nt * 16 + ln;
#pragma unroll
            for (int j = 0; j < 8; ++j) {
                float s = sacc[nt][j] * scale;
                const int rgl = rowm + j + 8 * half;
                if (cgl > rgl) s = -3.0e38f;
                sacc[nt][j] = s;
                mnew[j] = fmaxf(mnew[j], s);
            }
        }
        // reduce row max across the 16 lanes of this half-wave
#pragma unroll
        for (int d = 1; d < 16; d <<= 1)
#pragma unroll
            for (int j = 0; j < 8; ++j)
                mnew[j] = fmaxf(mnew[j], __shfl_xor(mnew[j], d, 32));

        float factor[8], rsum[8];
#pragma unroll
        for (int j = 0; j < 8; ++j) {
            factor[j] = __expf(mrow[j] - mnew[j]);
            mrow[j]   = mnew[j];
            rsum[j]   = 0.0f;
        }
#pragma unroll
        for (int nt = 0; nt < 8; ++nt)
#pragma unroll
            for (int j = 0; j < 8; ++j) {
                const float p = __expf(sacc[nt][j] - mnew[j]);
                sacc[nt][j] = p;
                rsum[j] += p;
            }
#pragma unroll
        for (int d = 1; d < 16; d <<= 1)
#pragma unroll
            for (int j = 0; j < 8; ++j) rsum[j] += __shfl_xor(rsum[j], d, 32);
#pragma unroll
        for (int j = 0; j < 8; ++j) lrow[j] = lrow[j] * factor[j] + rsum[j];
#pragma unroll
        for (int dt = 0; dt < 8; ++dt)
#pragma unroll
            for (int j = 0; j < 8; ++j) acc[dt][j] *= factor[j];

        __syncthreads();   // all waves done reading lsK -> safe to overwrite with P

        // ---- C-layout -> A-layout via per-wave LDS slice (reuse lsK)
        __bf16* P = lsK + wave * 16 * 128;
#pragma unroll
        for (int nt = 0; nt < 8; ++nt)
#pragma unroll
            for (int j = 0; j < 8; ++j)
                P[(j + 8 * half) * 128 + nt * 16 + ln] = (__bf16)sacc[nt][j];

        // ---- ctx += P * V  (K = 128 kv positions, 4 k-steps of 32)
#pragma unroll
        for (int kc = 0; kc < 4; ++kc) {
            const v16bf ap = pack_frag(
                *(const v8bf*)(P + ln * 128 + kc * 32 + 16 * half),
                *(const v8bf*)(P + ln * 128 + kc * 32 + 16 * half + 8));
            v16bf bv[8];
#pragma unroll
            for (int dt = 0; dt < 8; ++dt)
                bv[dt] = *(const v16bf*)(lsVt + (dt * 16 + ln) * 128 + kc * 32 + 16 * half);
#pragma unroll
            for (int dt = 0; dt < 8; ++dt)
                acc[dt] = __builtin_amdgcn_wmma_f32_16x16x32_bf16(
                    false, ap, false, bv[dt], (short)0, acc[dt], false, false);
        }
    }

    // ---- epilogue: normalize and store ctx (bf16)
#pragma unroll
    for (int dt = 0; dt < 8; ++dt)
#pragma unroll
        for (int j = 0; j < 8; ++j) {
            const int s = rowm + j + 8 * half;
            const float o = acc[dt][j] / lrow[j];
            ctx[((size_t)s * Bdim + b) * H2 + (size_t)h * DH + dt * 16 + ln] = (__bf16)o;
        }
}

// ---------------------------------------------------------------------------
extern "C" void kernel_launch(void* const* d_in, const int* in_sizes, int n_in,
                              void* d_out, int out_size, void* d_ws, size_t ws_size,
                              hipStream_t stream)
{
    (void)in_sizes; (void)n_in; (void)out_size; (void)ws_size;
    const float* x     = (const float*)d_in[0];   // [S,B,H2]
    const float* Wqkv  = (const float*)d_in[1];   // [3*H2, H2]
    const float* Wproj = (const float*)d_in[2];   // [HID, H2]
    float* out = (float*)d_out;                   // [S,B,HID]

    // workspace layout (bytes)
    char* ws = (char*)d_ws;
    __bf16* xb     = (__bf16*)(ws);                                    // 32 MiB
    __bf16* wqkvb  = (__bf16*)(ws + (size_t)32  * 1024 * 1024);        // 96 MiB
    __bf16* wprojb = (__bf16*)(ws + (size_t)128 * 1024 * 1024);        // 16 MiB
    __bf16* qkvb   = (__bf16*)(ws + (size_t)144 * 1024 * 1024);        // 96 MiB
    __bf16* ctxb   = (__bf16*)(ws + (size_t)240 * 1024 * 1024);        // 32 MiB

    // 1) fp32 -> bf16 conversions
    f32_to_bf16_kernel<<<2048, 256, 0, stream>>>(x,     xb,     (size_t)MROWS * H2);
    f32_to_bf16_kernel<<<2048, 256, 0, stream>>>(Wqkv,  wqkvb,  (size_t)O3 * H2);
    f32_to_bf16_kernel<<<2048, 256, 0, stream>>>(Wproj, wprojb, (size_t)HID * H2);

    // 2) QKV projection: [4096,4096] x [12288,4096]^T -> bf16 [4096,12288]
    gemm_bf16_wmma<true><<<dim3(O3 / 128, MROWS / 128), 256, 0, stream>>>(
        xb, wqkvb, (void*)qkvb, MROWS, O3, H2);

    // 3) causal flash attention -> bf16 ctx [4096,4096]
    attn_flash<<<dim3(Sdim / 128, NH, Bdim), 256, 0, stream>>>(qkvb, ctxb);

    // 4) output projection: [4096,4096] x [2048,4096]^T -> f32 [4096,2048]
    gemm_bf16_wmma<false><<<dim3(HID / 128, MROWS / 128), 256, 0, stream>>>(
        ctxb, wprojb, (void*)out, MROWS, HID, H2);
}
